// PhysioNetGeoLIF_4Class_34265249087579
// MI455X (gfx1250) — matrix operand order repaired and checked by reading, not compiled, not run on hardware
//
#include <hip/hip_runtime.h>

// ---- problem constants (match reference) ----
#define B_    256
#define T_    4096
#define C_    64
#define H_    64
#define TDA_  150
#define NC_   4

#define CHUNK     32    // timesteps staged per LDS tile
#define LDS_PITCH 72    // halves per row (64 + 8 pad) -> conflict-free b128 reads

typedef _Float16 v16h  __attribute__((ext_vector_type(16)));
typedef _Float16 half8 __attribute__((ext_vector_type(8)));
typedef _Float16 half4 __attribute__((ext_vector_type(4)));
typedef float    v8f   __attribute__((ext_vector_type(8)));

// Fused: spatial projection (WMMA f16 -> f32) + ReLU + sequential LIF scan.
// One workgroup per batch element b; wave w owns H columns [16w, 16w+16).
__global__ __launch_bounds__(128) void lif_scan_kernel(
    const float* __restrict__ kin,      // [B, T, C]
    const float* __restrict__ Wsp,      // [H, C]
    float* __restrict__ spike_sum)      // [B, H]
{
  __shared__ __align__(16) _Float16 tile[CHUNK * LDS_PITCH];

  const int b      = blockIdx.x;
  const int tid    = threadIdx.x;
  const int lane   = tid & 31;
  const int wave   = tid >> 5;       // 0..3
  const int lane16 = lane & 15;
  const bool hiHalf = lane >= 16;
  const int n0     = wave * 16;      // H-tile base for this wave

  // ---- resident WMMA B fragments: W_spatial as K x N (K=C, N=H tile) ----
  // 16-bit B 32x16 layout: low lanes K=0..7 & 16..23, high lanes K=8..15 & 24..31,
  // element e of the v16h is K = koffB + (e<8 ? e : e+8), with N = n0 + lane16.
  const float* wr = Wsp + (size_t)(n0 + lane16) * C_ + (hiHalf ? 8 : 0);
  v16h bf0, bf1;                     // K 0..31 and K 32..63
#pragma unroll
  for (int e = 0; e < 16; ++e) {
    bf0[e] = (_Float16)wr[e < 8 ? e : e + 8];
    bf1[e] = (_Float16)wr[(e < 8 ? e : e + 8) + 32];
  }

  const float* kb  = kin + (size_t)b * (T_ * C_);
  const int wrow = tid >> 2;          // 0..31  (timestep row in tile)
  const int wcol = (tid & 3) * 16;    // 0,16,32,48 (channel base)

  float carry  = 0.f;   // LIF membrane carry (valid in low lanes at tile entry)
  float spikes = 0.f;   // per half-lane partial spike count

  for (int tc = 0; tc < T_; tc += CHUNK) {
    // ---- stage CHUNK x C fp32 -> f16 LDS tile (coalesced b128 loads) ----
    const float* src = kb + (size_t)(tc + wrow) * C_ + wcol;
#pragma unroll
    for (int j = 0; j < 4; ++j) {
      float4 v = ((const float4*)src)[j];
      half4 hv = { (_Float16)v.x, (_Float16)v.y, (_Float16)v.z, (_Float16)v.w };
      *(half4*)&tile[wrow * LDS_PITCH + wcol + 4 * j] = hv;
    }
    if (tc + CHUNK < T_)  // keep HBM streaming ahead -> global_prefetch_b8
      __builtin_prefetch(kb + (size_t)(tc + CHUNK + wrow) * C_ + wcol, 0, 1);
    __syncthreads();

#pragma unroll
    for (int mt = 0; mt < CHUNK / 16; ++mt) {
      // ---- A fragments from LDS, 16-bit A 16x32 layout ----
      // lane row M = lane16; low lanes K=koffA+0..7 & +16..23, high lanes koffA=8.
      const int koffA = hiHalf ? 8 : 0;
      const _Float16* ap = &tile[(mt * 16 + lane16) * LDS_PITCH + koffA];
      half8 a00 = *(const half8*)(ap);        // K koffA + 0..7
      half8 a01 = *(const half8*)(ap + 16);   // K koffA + 16..23
      half8 a10 = *(const half8*)(ap + 32);   // K koffA + 32..39
      half8 a11 = *(const half8*)(ap + 48);   // K koffA + 48..55
      v16h a0, a1;
#pragma unroll
      for (int e = 0; e < 8; ++e) {
        a0[e] = a00[e]; a0[e + 8] = a01[e];
        a1[e] = a10[e]; a1[e + 8] = a11[e];
      }

      v8f acc = {};
      acc = __builtin_amdgcn_wmma_f32_16x16x32_f16(false, a0, false, bf0,
                                                   (short)0, acc, false, false);
      acc = __builtin_amdgcn_wmma_f32_16x16x32_f16(false, a1, false, bf1,
                                                   (short)0, acc, false, false);

      // ReLU. C/D layout: lane holds column n0+lane16; acc[i] is timestep
      // M = i (low lanes) or M = 8+i (high lanes).
      float vals[8];
#pragma unroll
      for (int i = 0; i < 8; ++i) vals[i] = acc[i] > 0.f ? acc[i] : 0.f;

      // ---- sequential LIF scan, split across lane halves ----
      float c = carry;
      float s = 0.f;
      if (!hiHalf) {               // M = 0..7
#pragma unroll
        for (int i = 0; i < 8; ++i) {
          c += vals[i];
          float spk = (c >= 1.f) ? 1.f : 0.f;
          s += spk;
          c *= (1.f - spk);        // hard reset
        }
      }
      float cmid = __shfl(c, lane16, 32);      // carry after M=7 -> high lanes
      if (hiHalf) {                // M = 8..15
        c = cmid;
#pragma unroll
        for (int i = 0; i < 8; ++i) {
          c += vals[i];
          float spk = (c >= 1.f) ? 1.f : 0.f;
          s += spk;
          c *= (1.f - spk);
        }
      }
      carry = __shfl(c, lane16 + 16, 32);      // carry after M=15 -> everyone
      spikes += s;
    }
    __syncthreads();   // tile consumed; safe to overwrite next iteration
  }

  float hs = __shfl(spikes, lane16 + 16, 32);  // high partner's count
  if (!hiHalf) spike_sum[b * H_ + n0 + lane16] = spikes + hs;
}

// Head: tda_weight = sigmoid(tda @ W_tda^T + b_tda); out = (spike*w) @ W_out^T + b_out
__global__ __launch_bounds__(64) void head_kernel(
    const float* __restrict__ tda,      // [B, TDA]
    const float* __restrict__ Wtda,     // [H, TDA]
    const float* __restrict__ btda,     // [H]
    const float* __restrict__ Wout,     // [NC, H]
    const float* __restrict__ bout,     // [NC]
    const float* __restrict__ spike_sum,// [B, H]
    float* __restrict__ out)            // [B, NC]
{
  __shared__ float gs[H_];
  const int b = blockIdx.x;
  const int h = threadIdx.x;

  float acc = btda[h];
  const float* tb = tda + (size_t)b * TDA_;
  const float* wrow = Wtda + (size_t)h * TDA_;
#pragma unroll 2
  for (int j = 0; j < TDA_; ++j) acc += tb[j] * wrow[j];
  float w = 1.f / (1.f + __expf(-acc));
  gs[h] = spike_sum[b * H_ + h] * w;
  __syncthreads();

  if (h < NC_) {
    float o = bout[h];
    const float* wo = Wout + h * H_;
#pragma unroll
    for (int j = 0; j < H_; ++j) o += wo[j] * gs[j];
    out[b * NC_ + h] = o;
  }
}

extern "C" void kernel_launch(void* const* d_in, const int* in_sizes, int n_in,
                              void* d_out, int out_size, void* d_ws, size_t ws_size,
                              hipStream_t stream) {
  (void)in_sizes; (void)n_in; (void)out_size; (void)d_ws; (void)ws_size;
  const float* kin  = (const float*)d_in[0];  // [B,T,C]
  const float* tda  = (const float*)d_in[1];  // [B,TDA]
  const float* Wsp  = (const float*)d_in[2];  // [H,C]
  const float* Wtda = (const float*)d_in[3];  // [H,TDA]
  const float* btda = (const float*)d_in[4];  // [H]
  const float* Wout = (const float*)d_in[5];  // [NC,H]
  const float* bout = (const float*)d_in[6];  // [NC]

  float* out   = (float*)d_out;               // [B,NC] first
  float* spike = out + B_ * NC_;              // then [B,H]

  lif_scan_kernel<<<B_, 128, 0, stream>>>(kin, Wsp, spike);
  head_kernel<<<B_, 64, 0, stream>>>(tda, Wtda, btda, Wout, bout, spike, out);
}
